// RandomBatchTripletLoss_ImgSearch_86612310491997
// MI455X (gfx1250) — matrix operand order repaired and checked by reading, not compiled, not run on hardware
//
#include <hip/hip_runtime.h>
#include <math.h>

#define BSZ 4096
#define DK  512
#define NIMG 8

typedef __attribute__((ext_vector_type(16))) __bf16 v16bf;
typedef __attribute__((ext_vector_type(8)))  __bf16 v8bf;
typedef __attribute__((ext_vector_type(4)))  __bf16 v4bf;
typedef __attribute__((ext_vector_type(8)))  float  v8f;

__device__ __forceinline__ __bf16 f32_to_bf16(float f) {
    unsigned u = __builtin_bit_cast(unsigned, f);
    unsigned r = (u + 0x7FFFu + ((u >> 16) & 1u)) >> 16;
    return __builtin_bit_cast(__bf16, (unsigned short)r);
}

__device__ __forceinline__ unsigned wang_hash(unsigned s) {
    s = (s ^ 61u) ^ (s >> 16);
    s *= 9u;
    s ^= s >> 4;
    s *= 0x27d4eb2du;
    s ^= s >> 15;
    return s;
}

__device__ __forceinline__ unsigned long long u64min(unsigned long long a,
                                                     unsigned long long b) {
    return a < b ? a : b;   // lowers to v_min_u64 on gfx1250
}

// ---------------------------------------------------------------------------
// Kernel 0: one-time f32 -> bf16 conversion of Q and G (removes all
// conversion VALU from the GEMM hot loop; 8 MB total, L2-resident).
// ---------------------------------------------------------------------------
__global__ __launch_bounds__(256)
void convert_bf16_kernel(const float* __restrict__ q, const float* __restrict__ g,
                         __bf16* __restrict__ qb, __bf16* __restrict__ gb) {
    const size_t total4 = (size_t)BSZ * DK / 4;     // float4 chunks per matrix
    size_t idx = (size_t)blockIdx.x * 256 + threadIdx.x;
    if (idx >= 2 * total4) return;
    const float4* src; __bf16* dst; size_t i4;
    if (idx < total4) { src = (const float4*)q; dst = qb; i4 = idx; }
    else              { src = (const float4*)g; dst = gb; i4 = idx - total4; }
    float4 v = src[i4];
    v4bf o;
    o[0] = f32_to_bf16(v.x); o[1] = f32_to_bf16(v.y);
    o[2] = f32_to_bf16(v.z); o[3] = f32_to_bf16(v.w);
    *(v4bf*)(dst + i4 * 4) = o;
}

// ---------------------------------------------------------------------------
// Kernel 1: row norms (f32), one wave32 per row.
// ---------------------------------------------------------------------------
__global__ __launch_bounds__(256)
void norms_kernel(const float* __restrict__ q, const float* __restrict__ g,
                  float* __restrict__ nq, float* __restrict__ ng) {
    const int wave = threadIdx.x >> 5;
    const int lane = threadIdx.x & 31;
    const int row  = blockIdx.x * 8 + wave;
    const float* src = (row < BSZ) ? &q[(size_t)row * DK]
                                   : &g[(size_t)(row - BSZ) * DK];
    float s = 0.f;
    for (int k = lane; k < DK; k += 32) { float v = src[k]; s += v * v; }
    #pragma unroll
    for (int off = 16; off; off >>= 1) s += __shfl_xor(s, off, 32);
    if (lane == 0) {
        if (row < BSZ) nq[row] = s;
        else           ng[row - BSZ] = s;
    }
}

// ---------------------------------------------------------------------------
// Kernel 2: positive sampling (7 in-block peers, f32 dots, categorical).
// ---------------------------------------------------------------------------
__global__ __launch_bounds__(256)
void pos_sample_kernel(const float* __restrict__ q, const float* __restrict__ g,
                       const float* __restrict__ nq, const float* __restrict__ ng,
                       float* __restrict__ pscore, float* __restrict__ spos) {
    const int wave = threadIdx.x >> 5;
    const int lane = threadIdx.x & 31;
    const int i    = blockIdx.x * 8 + wave;
    const int base = (i >> 3) << 3;
    const float* qi = &q[(size_t)i * DK];

    float dp[NIMG];
    #pragma unroll
    for (int j = 0; j < NIMG; ++j) {
        const float* gj = &g[(size_t)(base + j) * DK];
        float s = 0.f;
        for (int k = lane; k < DK; k += 32) s += qi[k] * gj[k];
        #pragma unroll
        for (int off = 16; off; off >>= 1) s += __shfl_xor(s, off, 32);
        float d = fmaxf(nq[i] + ng[base + j] - 2.f * s, 1e-8f);
        dp[j] = (base + j == i) ? 0.f : d;
    }
    if (lane == 0) {
        float total = 0.f;
        #pragma unroll
        for (int j = 0; j < NIMG; ++j) total += dp[j];
        unsigned h = wang_hash(0x9E3779B9u ^ (unsigned)i);
        float u = (h >> 8) * (1.0f / 16777216.0f) * total;
        float c = 0.f; int sel = -1;
        #pragma unroll
        for (int j = 0; j < NIMG; ++j) {
            if (dp[j] > 0.f) { c += dp[j]; if (sel < 0 && u <= c) sel = j; }
        }
        if (sel < 0) {
            for (int j = NIMG - 1; j >= 0; --j) if (dp[j] > 0.f) { sel = j; break; }
        }
        float ps = (sel >= 0) ? dp[sel] : 1e-8f;
        pscore[i] = ps;
        spos[i]   = sqrtf(ps);
    }
}

// ---------------------------------------------------------------------------
// Kernel 3: fused bf16 WMMA GEMM (Q.G^T) + distance epilogue + per-row
// negative mining. 16-row stripe per workgroup; 4 waves split N=4096.
// Epilogue is fully branchless (v_cndmask + v_min_u64); all stats in
// registers, merged into LDS atomics once at kernel end.
// ---------------------------------------------------------------------------
__global__ __launch_bounds__(128)
void dist_stats_kernel(const __bf16* __restrict__ qb, const __bf16* __restrict__ gb,
                       const float* __restrict__ nq, const float* __restrict__ ng,
                       const float* __restrict__ pscore,
                       unsigned* __restrict__ cnt_g,
                       unsigned long long* __restrict__ minkey_g,
                       unsigned long long* __restrict__ hashkey_g) {
    __shared__ __attribute__((aligned(32))) __bf16 sA[16][DK];   // 16 KB
    __shared__ unsigned s_cnt[16];
    __shared__ unsigned long long s_min[16];
    __shared__ unsigned long long s_hash[16];

    const int tid      = threadIdx.x;
    const int row_base = blockIdx.x * 16;

    if (tid < 16) {
        s_cnt[tid]  = 0u;
        s_min[tid]  = ~0ull;
        s_hash[tid] = ~0ull;
    }
    // Raw 16 KB copy of the pre-converted bf16 A stripe into LDS.
    {
        const uint4* src = (const uint4*)(qb + (size_t)row_base * DK);
        uint4* dst = (uint4*)&sA[0][0];
        for (int t = tid; t < 16 * DK / 8; t += 128) dst[t] = src[t];
    }
    __syncthreads();

    const int wave = tid >> 5;
    const int lane = tid & 31;
    const int mlo  = lane & 15;
    const bool hi  = (lane & 16) != 0;

    // Hoist per-row constants into registers (loaded once, broadcast).
    float rnq[8], rthr[8];
    int rowblk[8];
    unsigned rowmix[8];
    #pragma unroll
    for (int r = 0; r < 8; ++r) {
        const int row = row_base + r + (hi ? 8 : 0);
        rnq[r]    = nq[row];
        rthr[r]   = pscore[row] + 1e-4f;
        rowblk[r] = row >> 3;
        rowmix[r] = wang_hash((unsigned)row);
    }

    // Register-resident per-row stats for this lane's column subset.
    unsigned long long minloc[8], hashloc[8];
    unsigned cntloc[8];
    #pragma unroll
    for (int r = 0; r < 8; ++r) { minloc[r] = ~0ull; hashloc[r] = ~0ull; cntloc[r] = 0u; }

    const int n_end = wave * 1024 + 1024;
    for (int n0 = wave * 1024; n0 < n_end; n0 += 16) {
        const int n = n0 + mlo;
        if (n0 + 16 < n_end)
            __builtin_prefetch(&gb[(size_t)(n + 16) * DK], 0, 0);

        v8f acc = {};
        #pragma unroll 4
        for (int kk = 0; kk < DK; kk += 32) {
            // A fragment: lane m=mlo,
            // lo half-wave K = kk+0..15, hi = kk+16..31.
            const int kbA = kk + (hi ? 16 : 0);
            v8bf alo = *(const v8bf*)&sA[mlo][kbA];
            v8bf ahi = *(const v8bf*)&sA[mlo][kbA + 8];
            v16bf a = __builtin_shufflevector(alo, ahi,
                0, 1, 2, 3, 4, 5, 6, 7, 8, 9, 10, 11, 12, 13, 14, 15);
            // B fragment: lane n=mlo; lo half-wave K=kk+0..15, hi K=kk+16..31.
            // B[k][n] = G[n][k] -> one aligned 32B contiguous load.
            const int kbB = kk + (hi ? 16 : 0);
            v16bf b = *(const v16bf*)&gb[(size_t)n * DK + kbB];

            acc = __builtin_amdgcn_wmma_f32_16x16x32_bf16(
                false, a, false, b, (short)0, acc, false, false);
        }

        // Fused epilogue: fully branchless distance + mining updates.
        const float ngv    = ng[n];
        const int   nblk   = n >> 3;
        const unsigned nmx = (unsigned)n * 2654435761u;
        #pragma unroll
        for (int r = 0; r < 8; ++r) {
            float dist = fmaxf(rnq[r] + ngv - 2.0f * acc[r], 1e-8f);
            const bool isneg = (nblk != rowblk[r]);
            const bool qual  = isneg & (dist < rthr[r]);

            unsigned fb = __builtin_bit_cast(unsigned, dist); // dist>0 => ordered
            unsigned long long mk =
                isneg ? (((unsigned long long)fb << 32) | (unsigned)n) : ~0ull;
            minloc[r] = u64min(minloc[r], mk);

            cntloc[r] += (unsigned)qual;

            unsigned h = (rowmix[r] ^ nmx) * 0x9E3779B1u;
            unsigned long long hk =
                qual ? (((unsigned long long)h << 32) | (unsigned)n) : ~0ull;
            hashloc[r] = u64min(hashloc[r], hk);
        }
    }

    // Merge lane-local stats into per-row LDS slots (once).
    #pragma unroll
    for (int r = 0; r < 8; ++r) {
        const int m = r + (hi ? 8 : 0);
        atomicMin(&s_min[m], minloc[r]);
        atomicAdd(&s_cnt[m], cntloc[r]);
        atomicMin(&s_hash[m], hashloc[r]);
    }
    __syncthreads();
    if (tid < 16) {
        cnt_g[row_base + tid]     = s_cnt[tid];
        minkey_g[row_base + tid]  = s_min[tid];
        hashkey_g[row_base + tid] = s_hash[tid];
    }
}

// ---------------------------------------------------------------------------
// Kernel 4: per-row negative score (f32 recompute), softplus, mean.
// ---------------------------------------------------------------------------
__global__ __launch_bounds__(256)
void finalize_kernel(const float* __restrict__ q, const float* __restrict__ g,
                     const float* __restrict__ nq, const float* __restrict__ ng,
                     const float* __restrict__ spos,
                     const unsigned* __restrict__ cnt,
                     const unsigned long long* __restrict__ minkey,
                     const unsigned long long* __restrict__ hashkey,
                     float* __restrict__ out) {
    __shared__ float red[256];
    const int i = blockIdx.x * 256 + threadIdx.x;
    float sn;
    if (cnt[i] > 0u) {
        unsigned n = (unsigned)(hashkey[i] & 0xFFFFFFFFull);
        const float* qi = &q[(size_t)i * DK];
        const float* gn = &g[(size_t)n * DK];
        float s = 0.f;
        for (int k = 0; k < DK; ++k) s += qi[k] * gn[k];
        sn = sqrtf(fmaxf(nq[i] + ng[n] - 2.f * s, 1e-8f));
    } else {
        unsigned fb = (unsigned)(minkey[i] >> 32);
        sn = sqrtf(__builtin_bit_cast(float, fb));
    }
    float x = 1e-4f + spos[i] - sn;
    float l = (x > 20.f) ? x : log1pf(expf(x));
    red[threadIdx.x] = l;
    __syncthreads();
    #pragma unroll
    for (int off = 128; off; off >>= 1) {
        if (threadIdx.x < off) red[threadIdx.x] += red[threadIdx.x + off];
        __syncthreads();
    }
    if (threadIdx.x == 0) atomicAdd(out, red[0] * (1.0f / BSZ));
}

// ---------------------------------------------------------------------------
extern "C" void kernel_launch(void* const* d_in, const int* in_sizes, int n_in,
                              void* d_out, int out_size, void* d_ws, size_t ws_size,
                              hipStream_t stream) {
    (void)in_sizes; (void)n_in; (void)out_size; (void)ws_size;
    const float* q = (const float*)d_in[0];
    const float* g = (const float*)d_in[1];
    // d_in[2] = batch_size_image scalar (fixed at 8, baked into kernels)

    char* ws = (char*)d_ws;
    float*              nq      = (float*)(ws + 0);
    float*              ng      = (float*)(ws + 16 * 1024);
    float*              pscore  = (float*)(ws + 32 * 1024);
    float*              spos    = (float*)(ws + 48 * 1024);
    unsigned*           cnt     = (unsigned*)(ws + 64 * 1024);
    unsigned long long* minkey  = (unsigned long long*)(ws + 80 * 1024);
    unsigned long long* hashkey = (unsigned long long*)(ws + 112 * 1024);
    __bf16*             qb      = (__bf16*)(ws + 256 * 1024);
    __bf16*             gb      = (__bf16*)(ws + 256 * 1024 + (size_t)BSZ * DK * 2);
    float* out = (float*)d_out;

    hipMemsetAsync(out, 0, sizeof(float), stream);

    convert_bf16_kernel<<<4096, 256, 0, stream>>>(q, g, qb, gb);
    norms_kernel<<<1024, 256, 0, stream>>>(q, g, nq, ng);
    pos_sample_kernel<<<512, 256, 0, stream>>>(q, g, nq, ng, pscore, spos);
    dist_stats_kernel<<<256, 128, 0, stream>>>(qb, gb, nq, ng, pscore,
                                               cnt, minkey, hashkey);
    finalize_kernel<<<16, 256, 0, stream>>>(q, g, nq, ng, spos,
                                            cnt, minkey, hashkey, out);
}
